// TransformerDecoder_80229989089970
// MI455X (gfx1250) — matrix operand order repaired and checked by reading, not compiled, not run on hardware
//
#include <hip/hip_runtime.h>

// ---------------------------------------------------------------------------
// Types
// ---------------------------------------------------------------------------
typedef __bf16 bf16;
typedef __attribute__((ext_vector_type(16))) __bf16 v16bf;
typedef __attribute__((ext_vector_type(8)))  float  v8f;

#define B_  2
#define L_  2048
#define D_  1024
#define F_  4096
#define H_  16
#define HD_ 64
#define M_  (B_ * L_)   // 4096 tokens total

__device__ __forceinline__ bf16 to_bf16(float f) {
    unsigned u = __builtin_bit_cast(unsigned, f);
    unsigned r = (u + 0x7FFFu + ((u >> 16) & 1u)) >> 16;
    unsigned short h = (unsigned short)r;
    return __builtin_bit_cast(bf16, h);
}

// ---------------------------------------------------------------------------
// Elementwise f32 -> bf16 convert
// ---------------------------------------------------------------------------
__global__ __launch_bounds__(256)
void cvt_f32_bf16(const float* __restrict__ in, bf16* __restrict__ out, size_t n) {
    size_t i = (size_t)blockIdx.x * blockDim.x + threadIdx.x;
    size_t stride = (size_t)gridDim.x * blockDim.x;
    for (; i < n; i += stride) out[i] = to_bf16(in[i]);
}

// Transposing convert: in f32 [R][C] -> out bf16 [C][R]
__global__ __launch_bounds__(256)
void transpose_cvt(const float* __restrict__ in, bf16* __restrict__ out, int R, int C) {
    size_t n = (size_t)R * C;
    size_t i = (size_t)blockIdx.x * blockDim.x + threadIdx.x;
    size_t stride = (size_t)gridDim.x * blockDim.x;
    for (; i < n; i += stride) {
        int r = (int)(i / (size_t)C);
        int c = (int)(i % (size_t)C);
        out[(size_t)c * R + r] = to_bf16(in[i]);
    }
}

// ---------------------------------------------------------------------------
// bf16 WMMA GEMM: C[M][N] = A[M][K] @ Bt[N][K]^T (+bias)(+resid)(relu?)
// Block tile 128x128, 8 waves (wave32), wave tile 64x32 (4x2 WMMA 16x16 accums)
// K-step 32 staged through LDS with CDNA5 async global->LDS copies (ASYNCcnt).
// Epilogue writes f32 / bf16 / transposed bf16.
// ---------------------------------------------------------------------------
__global__ __launch_bounds__(256)
void gemm_bf16_kernel(const bf16* __restrict__ A, const bf16* __restrict__ Bt,
                      const float* __restrict__ bias, const float* __restrict__ resid,
                      float* __restrict__ outF, bf16* __restrict__ outB, bf16* __restrict__ outBT,
                      int M, int N, int K, int relu)
{
    __shared__ bf16 sA[128][48];   // row stride 96B -> b128-aligned
    __shared__ bf16 sB[128][48];

    const int tid  = threadIdx.x;
    const int wid  = tid >> 5;
    const int lane = tid & 31;
    const int lr   = lane & 15;
    const int half = lane >> 4;            // 0: lanes 0-15, 1: lanes 16-31
    const int bm   = blockIdx.x * 128;
    const int bn   = blockIdx.y * 128;
    const int wm   = (wid >> 2) * 64;      // 0 / 64
    const int wn   = (wid & 3) * 32;       // 0..96

    v8f acc[4][2];
    #pragma unroll
    for (int mi = 0; mi < 4; ++mi)
        #pragma unroll
        for (int ni = 0; ni < 2; ++ni)
            #pragma unroll
            for (int r = 0; r < 8; ++r) acc[mi][ni][r] = 0.f;

    union AF { v16bf v; int4 q[2]; };

    // Per-thread staging coordinates: each thread moves 2x 16B per tile.
    const int idx0 = tid * 8;               // elements 0..2047
    const int r0   = idx0 >> 5;             // rows 0..63
    const int c0   = idx0 & 31;
    const unsigned ldsA0 = (unsigned)(uintptr_t)&sA[r0][c0];
    const unsigned ldsA1 = (unsigned)(uintptr_t)&sA[r0 + 64][c0];
    const unsigned ldsB0 = (unsigned)(uintptr_t)&sB[r0][c0];
    const unsigned ldsB1 = (unsigned)(uintptr_t)&sB[r0 + 64][c0];

    for (int k0 = 0; k0 < K; k0 += 32) {
        // ---- async stage A[128][32] and Bt[128][32] tiles directly into LDS
        {
            const bf16* gA0 = &A [(size_t)(bm + r0)      * K + k0 + c0];
            const bf16* gA1 = &A [(size_t)(bm + r0 + 64) * K + k0 + c0];
            const bf16* gB0 = &Bt[(size_t)(bn + r0)      * K + k0 + c0];
            const bf16* gB1 = &Bt[(size_t)(bn + r0 + 64) * K + k0 + c0];
            asm volatile("global_load_async_to_lds_b128 %0, %1, off"
                         :: "v"(ldsA0), "v"(gA0) : "memory");
            asm volatile("global_load_async_to_lds_b128 %0, %1, off"
                         :: "v"(ldsA1), "v"(gA1) : "memory");
            asm volatile("global_load_async_to_lds_b128 %0, %1, off"
                         :: "v"(ldsB0), "v"(gB0) : "memory");
            asm volatile("global_load_async_to_lds_b128 %0, %1, off"
                         :: "v"(ldsB1), "v"(gB1) : "memory");
            asm volatile("s_wait_asynccnt 0x0" ::: "memory");
        }
        __syncthreads();

        AF aF[4], bF[2];
        #pragma unroll
        for (int mi = 0; mi < 4; ++mi) {
            // A-frag 16x32: lanes<16 K0..15, lanes>=16 K16..31 (contiguous)
            const bf16* p = &sA[wm + mi * 16 + lr][half * 16];
            aF[mi].q[0] = *(const int4*)p;
            aF[mi].q[1] = *(const int4*)(p + 8);
        }
        #pragma unroll
        for (int ni = 0; ni < 2; ++ni) {
            // B-frag 32x16: lanes<16 K0..15, lanes>=16 K16..31 (contiguous)
            const bf16* p = &sB[wn + ni * 16 + lr][half * 16];
            bF[ni].q[0] = *(const int4*)p;
            bF[ni].q[1] = *(const int4*)(p + 8);
        }
        #pragma unroll
        for (int mi = 0; mi < 4; ++mi)
            #pragma unroll
            for (int ni = 0; ni < 2; ++ni)
                acc[mi][ni] = __builtin_amdgcn_wmma_f32_16x16x32_bf16(
                    false, aF[mi].v, false, bF[ni].v, (short)0, acc[mi][ni], false, false);
        __syncthreads();
    }

    // epilogue: C-layout row = vgpr + 8*half, col = lane&15
    #pragma unroll
    for (int mi = 0; mi < 4; ++mi)
        #pragma unroll
        for (int ni = 0; ni < 2; ++ni)
            #pragma unroll
            for (int r = 0; r < 8; ++r) {
                int gm = bm + wm + mi * 16 + r + half * 8;
                int gn = bn + wn + ni * 16 + lr;
                float v = acc[mi][ni][r];
                if (bias)  v += bias[gn];
                if (resid) v += resid[(size_t)gm * N + gn];
                if (relu)  v = fmaxf(v, 0.f);
                if (outF)  outF[(size_t)gm * N + gn] = v;
                if (outB)  outB[(size_t)gm * N + gn] = to_bf16(v);
                if (outBT) outBT[(size_t)gn * M + gm] = to_bf16(v);
            }
}

// ---------------------------------------------------------------------------
// Flash-style MHA core. Q,K bf16 [B,L,D]; Vt bf16 [D][B*L] (channel-major);
// mask int32 [B,1,L,L]; O bf16 [B,L,D]. One block = 4 waves; each wave owns
// 16 query rows of one (b,h); loops over 32-key blocks with online softmax.
// ---------------------------------------------------------------------------
__global__ __launch_bounds__(128)
void attention_kernel(const bf16* __restrict__ Q, const bf16* __restrict__ K,
                      const bf16* __restrict__ Vt, const int* __restrict__ mask,
                      bf16* __restrict__ O)
{
    __shared__ float sS[4][16][33];
    __shared__ bf16  sP[4][16][48];
    __shared__ float sAl[4][16];
    __shared__ float sL[4][16];

    const int tid  = threadIdx.x;
    const int w    = tid >> 5;
    const int lane = tid & 31;
    const int lr   = lane & 15;
    const int half = lane >> 4;
    const int bh   = blockIdx.x;
    const int b    = bh / H_;
    const int h    = bh % H_;
    const int q0   = blockIdx.y * 64 + w * 16;

    union AF { v16bf v; int4 q[2]; };

    // Q fragments for 16 rows x 64 (two 16x32 A-frags)
    AF aQ[2];
    {
        size_t qrow = (size_t)(b * L_ + q0 + lr) * D_ + h * HD_;
        #pragma unroll
        for (int s = 0; s < 2; ++s) {
            const bf16* p = Q + qrow + s * 32 + half * 16;
            aQ[s].q[0] = *(const int4*)p;
            aQ[s].q[1] = *(const int4*)(p + 8);
        }
    }

    v8f o[4];
    #pragma unroll
    for (int ni = 0; ni < 4; ++ni)
        #pragma unroll
        for (int r = 0; r < 8; ++r) o[ni][r] = 0.f;

    float m_run = -3.0e38f, l_run = 0.f;
    const size_t mbase = ((size_t)b * L_ + (size_t)(q0 + lr)) * (size_t)L_;

    for (int kb = 0; kb < L_; kb += 32) {
        // ---- S = (Q @ K^T) * 1/sqrt(hd) for 32 keys (2 key tiles x 2 k-steps)
        v8f sc[2];
        #pragma unroll
        for (int j = 0; j < 2; ++j)
            #pragma unroll
            for (int r = 0; r < 8; ++r) sc[j][r] = 0.f;
        #pragma unroll
        for (int j = 0; j < 2; ++j)
            #pragma unroll
            for (int s = 0; s < 2; ++s) {
                AF bK;
                const bf16* p = K + (size_t)(b * L_ + kb + j * 16 + lr) * D_
                                  + h * HD_ + s * 32 + half * 16;
                bK.q[0] = *(const int4*)p;
                bK.q[1] = *(const int4*)(p + 8);
                sc[j] = __builtin_amdgcn_wmma_f32_16x16x32_bf16(
                    false, aQ[s].v, false, bK.v, (short)0, sc[j], false, false);
            }
        #pragma unroll
        for (int j = 0; j < 2; ++j)
            #pragma unroll
            for (int r = 0; r < 8; ++r)
                sS[w][r + half * 8][j * 16 + lr] = sc[j][r] * 0.125f;
        __syncthreads();

        // ---- online softmax: lane pair (lr, lr+16) owns row lr, splits keys
        {
            const int kh = half * 16;
            const int* mrow = mask + mbase + kb + kh;
            float bm = -3.0e38f;
            #pragma unroll
            for (int k = 0; k < 16; ++k) {
                float s = sS[w][lr][kh + k];
                if (mrow[k] == 0) s = -1.0e9f;
                sS[w][lr][kh + k] = s;
                bm = fmaxf(bm, s);
            }
            bm = fmaxf(bm, __shfl_xor(bm, 16, 32));
            float mn = fmaxf(m_run, bm);
            float alpha = __expf(m_run - mn);
            float sum = 0.f;
            #pragma unroll
            for (int k = 0; k < 16; ++k) {
                float p = __expf(sS[w][lr][kh + k] - mn);
                sum += p;
                sP[w][lr][kh + k] = to_bf16(p);
            }
            sum += __shfl_xor(sum, 16, 32);
            l_run = alpha * l_run + sum;
            m_run = mn;
            if (half == 0) sAl[w][lr] = alpha;
        }
        __syncthreads();

        // ---- rescale O accumulators by alpha(row)
        float al[8];
        #pragma unroll
        for (int r = 0; r < 8; ++r) al[r] = sAl[w][r + half * 8];
        #pragma unroll
        for (int ni = 0; ni < 4; ++ni)
            #pragma unroll
            for (int r = 0; r < 8; ++r) o[ni][r] *= al[r];

        // ---- O += P(16q x 32k) @ V(32k x 64hd)
        AF aP;
        {
            const bf16* p = &sP[w][lr][half * 16];
            aP.q[0] = *(const int4*)p;
            aP.q[1] = *(const int4*)(p + 8);
        }
        #pragma unroll
        for (int ni = 0; ni < 4; ++ni) {
            AF bV;
            const bf16* p = Vt + (size_t)(h * HD_ + ni * 16 + lr) * M_
                               + b * L_ + kb + half * 16;
            bV.q[0] = *(const int4*)p;
            bV.q[1] = *(const int4*)(p + 8);
            o[ni] = __builtin_amdgcn_wmma_f32_16x16x32_bf16(
                false, aP.v, false, bV.v, (short)0, o[ni], false, false);
        }
        __syncthreads();
    }

    if (half == 0) sL[w][lr] = l_run;
    __syncthreads();
    float linv[8];
    #pragma unroll
    for (int r = 0; r < 8; ++r) linv[r] = 1.f / sL[w][r + half * 8];
    #pragma unroll
    for (int ni = 0; ni < 4; ++ni)
        #pragma unroll
        for (int r = 0; r < 8; ++r) {
            size_t row = (size_t)(b * L_ + q0 + r + half * 8);
            O[row * D_ + h * HD_ + ni * 16 + lr] = to_bf16(o[ni][r] * linv[r]);
        }
}

// ---------------------------------------------------------------------------
// y = LN(x) * g + b ; writes f32 and optional bf16. One block per row (D=1024).
// ---------------------------------------------------------------------------
__global__ __launch_bounds__(256)
void ln_kernel(const float* __restrict__ x, const float* __restrict__ g,
               const float* __restrict__ beta,
               float* __restrict__ outF, bf16* __restrict__ outB)
{
    __shared__ float red[256];
    const int row = blockIdx.x;
    const int t = threadIdx.x;
    const float* xr = x + (size_t)row * D_;
    float4 v4 = *(const float4*)&xr[t * 4];
    float vv[4] = { v4.x, v4.y, v4.z, v4.w };

    red[t] = vv[0] + vv[1] + vv[2] + vv[3];
    __syncthreads();
    for (int off = 128; off > 0; off >>= 1) {
        if (t < off) red[t] += red[t + off];
        __syncthreads();
    }
    float mu = red[0] * (1.f / D_);
    __syncthreads();

    float sq = 0.f;
    #pragma unroll
    for (int e = 0; e < 4; ++e) { float d = vv[e] - mu; sq += d * d; }
    red[t] = sq;
    __syncthreads();
    for (int off = 128; off > 0; off >>= 1) {
        if (t < off) red[t] += red[t + off];
        __syncthreads();
    }
    float rstd = rsqrtf(red[0] * (1.f / D_) + 1e-5f);

    float* orow = outF + (size_t)row * D_;
    bf16*  brow = outB ? outB + (size_t)row * D_ : nullptr;
    #pragma unroll
    for (int e = 0; e < 4; ++e) {
        int col = t * 4 + e;
        float y = (vv[e] - mu) * rstd * g[col] + beta[col];
        orow[col] = y;
        if (brow) brow[col] = to_bf16(y);
    }
}

// ---------------------------------------------------------------------------
// Host-side orchestration
// ---------------------------------------------------------------------------
extern "C" void kernel_launch(void* const* d_in, const int* in_sizes, int n_in,
                              void* d_out, int out_size, void* d_ws, size_t ws_size,
                              hipStream_t stream)
{
    (void)in_sizes; (void)n_in; (void)out_size; (void)ws_size;

    const float* x    = (const float*)d_in[0];
    const float* enc  = (const float*)d_in[1];
    const int*   imsk = (const int*)d_in[2];
    const int*   amsk = (const int*)d_in[3];
    const float* saW  = (const float*)d_in[4];
    const float* sab  = (const float*)d_in[5];
    const float* caW  = (const float*)d_in[6];
    const float* cab  = (const float*)d_in[7];
    const float* W1   = (const float*)d_in[8];
    const float* b1   = (const float*)d_in[9];
    const float* W2   = (const float*)d_in[10];
    const float* b2   = (const float*)d_in[11];
    const float* lng  = (const float*)d_in[12];
    const float* lnb  = (const float*)d_in[13];
    float* out = (float*)d_out;

    char* wsp = (char*)d_ws;
    auto alloc = [&](size_t bytes) -> void* {
        void* p = (void*)wsp;
        wsp += (bytes + 255) & ~(size_t)255;
        return p;
    };

    bf16* xb   = (bf16*)alloc((size_t)M_ * D_ * 2);
    bf16* eb   = (bf16*)alloc((size_t)M_ * D_ * 2);
    bf16* qb   = (bf16*)alloc((size_t)M_ * D_ * 2);
    bf16* kbuf = (bf16*)alloc((size_t)M_ * D_ * 2);
    bf16* vtb  = (bf16*)alloc((size_t)M_ * D_ * 2);   // V transposed [D][M]
    bf16* aob  = (bf16*)alloc((size_t)M_ * D_ * 2);
    bf16* x1b  = (bf16*)alloc((size_t)M_ * D_ * 2);
    bf16* x2b  = (bf16*)alloc((size_t)M_ * D_ * 2);
    bf16* hb   = (bf16*)alloc((size_t)M_ * F_ * 2);
    float* sf  = (float*)alloc((size_t)M_ * D_ * 4);
    float* x1f = (float*)alloc((size_t)M_ * D_ * 4);
    float* x2f = (float*)alloc((size_t)M_ * D_ * 4);
    bf16* saWt = (bf16*)alloc((size_t)4 * D_ * D_ * 2);
    bf16* caWt = (bf16*)alloc((size_t)4 * D_ * D_ * 2);
    bf16* w1t  = (bf16*)alloc((size_t)D_ * F_ * 2);
    bf16* w2t  = (bf16*)alloc((size_t)F_ * D_ * 2);

    // ---- precision conversion + weight transposition
    cvt_f32_bf16<<<2048, 256, 0, stream>>>(x,   xb, (size_t)M_ * D_);
    cvt_f32_bf16<<<2048, 256, 0, stream>>>(enc, eb, (size_t)M_ * D_);
    for (int i = 0; i < 4; ++i) {
        transpose_cvt<<<2048, 256, 0, stream>>>(saW + (size_t)i * D_ * D_,
                                                saWt + (size_t)i * D_ * D_, D_, D_);
        transpose_cvt<<<2048, 256, 0, stream>>>(caW + (size_t)i * D_ * D_,
                                                caWt + (size_t)i * D_ * D_, D_, D_);
    }
    transpose_cvt<<<4096, 256, 0, stream>>>(W1, w1t, D_, F_);  // -> [F][D]
    transpose_cvt<<<4096, 256, 0, stream>>>(W2, w2t, F_, D_);  // -> [D][F]

    dim3 gProj(M_ / 128, D_ / 128);
    dim3 gFfn1(M_ / 128, F_ / 128);
    dim3 blk(256);
    dim3 gAttn(B_ * H_, L_ / 64);
    dim3 bAttn(128);

    // ---- masked self-attention + residual + LN
    gemm_bf16_kernel<<<gProj, blk, 0, stream>>>(xb, saWt + 0 * (size_t)D_ * D_, sab + 0 * D_,
                                                nullptr, nullptr, qb, nullptr, M_, D_, D_, 0);
    gemm_bf16_kernel<<<gProj, blk, 0, stream>>>(xb, saWt + 1 * (size_t)D_ * D_, sab + 1 * D_,
                                                nullptr, nullptr, kbuf, nullptr, M_, D_, D_, 0);
    gemm_bf16_kernel<<<gProj, blk, 0, stream>>>(xb, saWt + 2 * (size_t)D_ * D_, sab + 2 * D_,
                                                nullptr, nullptr, nullptr, vtb, M_, D_, D_, 0);
    attention_kernel<<<gAttn, bAttn, 0, stream>>>(qb, kbuf, vtb, imsk, aob);
    gemm_bf16_kernel<<<gProj, blk, 0, stream>>>(aob, saWt + 3 * (size_t)D_ * D_, sab + 3 * D_,
                                                x, sf, nullptr, nullptr, M_, D_, D_, 0);
    ln_kernel<<<M_, 256, 0, stream>>>(sf, lng + 0 * D_, lnb + 0 * D_, x1f, x1b);

    // ---- encoder cross-attention + residual + LN
    gemm_bf16_kernel<<<gProj, blk, 0, stream>>>(x1b, caWt + 0 * (size_t)D_ * D_, cab + 0 * D_,
                                                nullptr, nullptr, qb, nullptr, M_, D_, D_, 0);
    gemm_bf16_kernel<<<gProj, blk, 0, stream>>>(eb, caWt + 1 * (size_t)D_ * D_, cab + 1 * D_,
                                                nullptr, nullptr, kbuf, nullptr, M_, D_, D_, 0);
    gemm_bf16_kernel<<<gProj, blk, 0, stream>>>(eb, caWt + 2 * (size_t)D_ * D_, cab + 2 * D_,
                                                nullptr, nullptr, nullptr, vtb, M_, D_, D_, 0);
    attention_kernel<<<gAttn, bAttn, 0, stream>>>(qb, kbuf, vtb, amsk, aob);
    gemm_bf16_kernel<<<gProj, blk, 0, stream>>>(aob, caWt + 3 * (size_t)D_ * D_, cab + 3 * D_,
                                                x1f, sf, nullptr, nullptr, M_, D_, D_, 0);
    ln_kernel<<<M_, 256, 0, stream>>>(sf, lng + 1 * D_, lnb + 1 * D_, x2f, x2b);

    // ---- FFN + residual + LN
    gemm_bf16_kernel<<<gFfn1, blk, 0, stream>>>(x2b, w1t, b1,
                                                nullptr, nullptr, hb, nullptr, M_, F_, D_, 1);
    gemm_bf16_kernel<<<gProj, blk, 0, stream>>>(hb, w2t, b2,
                                                x2f, sf, nullptr, nullptr, M_, D_, F_, 0);
    ln_kernel<<<M_, 256, 0, stream>>>(sf, lng + 2 * D_, lnb + 2 * D_, out, nullptr);
}
